// LlamaAttention_23304492548220
// MI455X (gfx1250) — matrix operand order repaired and checked by reading, not compiled, not run on hardware
//
#include <hip/hip_runtime.h>

#define HIDDEN 2048
#define NHEADS 16
#define HDIM   128
#define WINDOW 256
#define SEQ    8192
#define NWIN   (SEQ / WINDOW)

typedef __bf16 bf16;
typedef __attribute__((ext_vector_type(4)))  __bf16 v4bf;
typedef __attribute__((ext_vector_type(8)))  __bf16 v8bf;
typedef __attribute__((ext_vector_type(16))) __bf16 v16bf;
typedef __attribute__((ext_vector_type(8)))  float  v8f;

__device__ __forceinline__ bf16 f2bf(float x) {
    unsigned u = __builtin_bit_cast(unsigned, x);
    unsigned r = u + 0x7FFFu + ((u >> 16) & 1u);   // round-to-nearest-even
    unsigned short h = (unsigned short)(r >> 16);
    return __builtin_bit_cast(bf16, h);
}

// Low 32 bits of a generic pointer into LDS == wave-relative LDS byte offset
// (shared aperture base has zero low bits per the flat-addressing rules).
__device__ __forceinline__ unsigned lds_off(const void* p) {
    return (unsigned)(unsigned long long)(uintptr_t)p;
}

// A fragment: 16(M) x 32(K), source row-major [row][k] (k contiguous).
// lane&15 -> M, lane>>4 -> ksel; element j -> K = k0 + ksel*8 + (j&7) + (j>>3)*16
__device__ __forceinline__ v16bf load_fragA(const bf16* base, int ld, int row0, int k0) {
    int lane = threadIdx.x & 31;
    int r = lane & 15, ks = lane >> 4;
    const bf16* p = base + (long)(row0 + r) * ld + (k0 + ks * 8);
    v8bf lo = *(const v8bf*)p;
    v8bf hi = *(const v8bf*)(p + 16);
    v16bf out;
#pragma unroll
    for (int i = 0; i < 8; ++i) { out[i] = lo[i]; out[i + 8] = hi[i]; }
    return out;
}

// B fragment: 32(K) x 16(N), source "N-major": base[(col)*ld + k] (k contiguous).
// lane&15 -> N, lane>>4 -> kgroup; element j -> K = k0 + kg*8 + (j&7) + (j>>3)*16
__device__ __forceinline__ v16bf load_fragB(const bf16* base, long ld, int col0, int k0) {
    int lane = threadIdx.x & 31;
    int n = lane & 15, kg = lane >> 4;
    const bf16* p = base + (long)(col0 + n) * ld + (k0 + kg * 8);
    v8bf lo = *(const v8bf*)p;
    v8bf hi = *(const v8bf*)(p + 16);
    v16bf out;
#pragma unroll
    for (int i = 0; i < 8; ++i) { out[i] = lo[i]; out[i + 8] = hi[i]; }
    return out;
}

__global__ void f32_to_bf16_kernel(const float* __restrict__ in, bf16* __restrict__ out, long n4) {
    long stride = (long)gridDim.x * blockDim.x;
    for (long i = (long)blockIdx.x * blockDim.x + threadIdx.x; i < n4; i += stride) {
        float4 v = ((const float4*)in)[i];
        v4bf o;
        o[0] = f2bf(v.x); o[1] = f2bf(v.y); o[2] = f2bf(v.z); o[3] = f2bf(v.w);
        ((v4bf*)out)[i] = o;
    }
}

// ---------------------------------------------------------------------------
// 128x128 GEMM macro-tile over K=HIDDEN with async global->LDS double buffering.
// Arows: [128][HIDDEN] activation rows, Brows: [128][HIDDEN] weight rows (N-major).
// Each thread stages 32B of A and 32B of B per K-tile via
// GLOBAL_LOAD_ASYNC_TO_LDS_B128 (ASYNCcnt), 2-stage pipeline.
// ---------------------------------------------------------------------------
__device__ __forceinline__ void gemm128_async(
    const bf16* __restrict__ Arows, const bf16* __restrict__ Brows,
    bf16* sA, bf16* sB, v8f acc[8])
{
    const int tid = threadIdx.x;
    const int wave = tid >> 5;
    const int lr = tid >> 1;           // 0..127: staged row
    const int lc = (tid & 1) * 16;     // 0 / 16: staged k-half
    const unsigned laBase = lds_off(sA + lr * 32 + lc);
    const unsigned lbBase = lds_off(sB + lr * 32 + lc);
    const bf16* ga = Arows + (long)lr * HIDDEN + lc;
    const bf16* gb = Brows + (long)lr * HIDDEN + lc;

    const int NT = HIDDEN / 32;        // 64 K-tiles

    // prologue: tile 0 -> stage 0
    {
        unsigned long long pa = (unsigned long long)ga;
        unsigned long long pb = (unsigned long long)gb;
        asm volatile(
            "global_load_async_to_lds_b128 %0, %2, off\n\t"
            "global_load_async_to_lds_b128 %0, %2, off offset:16\n\t"
            "global_load_async_to_lds_b128 %1, %3, off\n\t"
            "global_load_async_to_lds_b128 %1, %3, off offset:16\n\t"
            "s_wait_xcnt 0x0"
            :: "v"(laBase), "v"(lbBase), "v"(pa), "v"(pb) : "memory");
    }

    for (int it = 0; it < NT; ++it) {
        if (it + 1 < NT) {
            unsigned stage = (unsigned)((it + 1) & 1) * (128 * 32 * 2);
            unsigned la = laBase + stage;
            unsigned lb = lbBase + stage;
            unsigned long long pa = (unsigned long long)(ga + (it + 1) * 32);
            unsigned long long pb = (unsigned long long)(gb + (it + 1) * 32);
            asm volatile(
                "global_load_async_to_lds_b128 %0, %2, off\n\t"
                "global_load_async_to_lds_b128 %0, %2, off offset:16\n\t"
                "global_load_async_to_lds_b128 %1, %3, off\n\t"
                "global_load_async_to_lds_b128 %1, %3, off offset:16\n\t"
                "s_wait_xcnt 0x0"
                :: "v"(la), "v"(lb), "v"(pa), "v"(pb) : "memory");
            // 4 newest stay in flight; current tile's 4 are complete (in-order)
            asm volatile("s_wait_asynccnt 0x4" ::: "memory");
        } else {
            asm volatile("s_wait_asynccnt 0x0" ::: "memory");
        }
        __syncthreads();
        const bf16* bufA = sA + (it & 1) * (128 * 32);
        const bf16* bufB = sB + (it & 1) * (128 * 32);
        v16bf a = load_fragA(bufA, 32, wave * 16, 0);
#pragma unroll
        for (int t = 0; t < 8; ++t) {
            v16bf b = load_fragB(bufB, 32, t * 16, 0);
            acc[t] = __builtin_amdgcn_wmma_f32_16x16x32_bf16(false, a, false, b,
                                                             (short)0, acc[t], false, false);
        }
        __syncthreads();
    }
}

// QKV GEMM + RoPE/mask epilogue.  grid = (SEQ/128, NHEADS, 3)  block = 256 (8 waves)
// z=0 -> Q (rope) -> Qh[h][s][d]; z=1 -> K (rope) -> Kh[h][s][d]; z=2 -> V -> Vt[h][d][s]
__global__ __launch_bounds__(256) void qkv_gemm(
    const bf16* __restrict__ Xb, const bf16* __restrict__ Wb,  // Wb = Wq|Wk|Wv contiguous
    const int* __restrict__ msk, const int* __restrict__ pos,
    bf16* __restrict__ Qh, bf16* __restrict__ Kh, bf16* __restrict__ Vt)
{
    __shared__ bf16 sA[2 * 128 * 32];
    __shared__ bf16 sB[2 * 128 * 32];
    const int m0 = blockIdx.x * 128;
    const int head = blockIdx.y;
    const int n0 = head * HDIM;
    const int z = blockIdx.z;
    const int tid = threadIdx.x, wave = tid >> 5, lane = tid & 31;

    v8f acc[8];
#pragma unroll
    for (int t = 0; t < 8; ++t)
#pragma unroll
        for (int e = 0; e < 8; ++e) acc[t][e] = 0.f;

    gemm128_async(Xb + (long)m0 * HIDDEN,
                  Wb + (long)z * HIDDEN * HIDDEN + (long)n0 * HIDDEN,
                  sA, sB, acc);

    const int col = lane & 15, hi = lane >> 4;
    if (z == 2) {
        // V epilogue: mask + transposed store Vt[h][d][s] (8 contiguous bf16 per lane/tile)
#pragma unroll
        for (int t = 0; t < 8; ++t) {
            int d = t * 16 + col;
            v4bf p0, p1;
#pragma unroll
            for (int r = 0; r < 8; ++r) {
                int s = m0 + wave * 16 + hi * 8 + r;
                float v = msk[s] ? acc[t][r] : 0.f;
                if (r < 4) p0[r] = f2bf(v); else p1[r - 4] = f2bf(v);
            }
            bf16* dst = Vt + ((long)head * HDIM + d) * SEQ + (m0 + wave * 16 + hi * 8);
            *(v4bf*)dst = p0;
            *(v4bf*)(dst + 4) = p1;
        }
    } else {
        bf16* out = (z == 0) ? Qh : Kh;
        float invf[4];
#pragma unroll
        for (int q = 0; q < 4; ++q) {
            float fidx = (float)(q * 16 + col);                 // = d (< 64)
            invf[q] = __expf(fidx * (-2.0f / 128.0f) * 9.210340371976184f); // 10000^(-2d/128)
        }
#pragma unroll
        for (int r = 0; r < 8; ++r) {
            int s = m0 + wave * 16 + hi * 8 + r;
            int mk = msk[s];
            float local = (float)(pos[s] % WINDOW);
            float res[8];
#pragma unroll
            for (int q = 0; q < 4; ++q) {
                float ang = local * invf[q];
                float c = cosf(ang), sn = sinf(ang);
                float x1 = acc[q][r];        // d      (< 64)
                float x2 = acc[q + 4][r];    // d + 64
                res[q]     = x1 * c - x2 * sn;
                res[q + 4] = x2 * c + x1 * sn;
            }
#pragma unroll
            for (int t = 0; t < 8; ++t) {
                float v = mk ? res[t] : 0.f;
                out[((long)head * SEQ + s) * HDIM + (t * 16 + col)] = f2bf(v);
            }
        }
    }
}

// Windowed causal attention.  grid = (NWIN, NHEADS)  block = 256 (8 waves; 2 q-tiles/wave)
__global__ __launch_bounds__(256) void win_attn(
    const bf16* __restrict__ Qh, const bf16* __restrict__ Kh, const bf16* __restrict__ Vt,
    bf16* __restrict__ attn)
{
    __shared__ bf16 sP[8][16 * 32];
    const int win = blockIdx.x, head = blockIdx.y;
    const int tid = threadIdx.x, wave = tid >> 5, lane = tid & 31;
    const int col = lane & 15, hi = lane >> 4;
    const bf16* Qb = Qh + (long)head * SEQ * HDIM;
    const bf16* Kb = Kh + (long)head * SEQ * HDIM;
    const bf16* Vb = Vt + (long)head * HDIM * SEQ;
    const float sm = 0.08838834764831845f;   // 1/sqrt(128)

    for (int sub = 0; sub < 2; ++sub) {
        const int qt = wave * 2 + sub;
        const int qrow0 = qt * 16;
        v16bf qf[4];
#pragma unroll
        for (int dc = 0; dc < 4; ++dc)
            qf[dc] = load_fragA(Qb, HDIM, win * WINDOW + qrow0, dc * 32);

        v8f O[8];
#pragma unroll
        for (int t = 0; t < 8; ++t)
#pragma unroll
            for (int e = 0; e < 8; ++e) O[t][e] = 0.f;
        float mrow[8], lrow[8];
#pragma unroll
        for (int r = 0; r < 8; ++r) { mrow[r] = -1e30f; lrow[r] = 0.f; }

        for (int p = 0; p < 8; ++p) {
            const int kc0 = p * 32;
            v8f S0, S1;
#pragma unroll
            for (int e = 0; e < 8; ++e) { S0[e] = 0.f; S1[e] = 0.f; }
#pragma unroll
            for (int dc = 0; dc < 4; ++dc) {
                v16bf b0 = load_fragB(Kb, HDIM, win * WINDOW + kc0,      dc * 32);
                v16bf b1 = load_fragB(Kb, HDIM, win * WINDOW + kc0 + 16, dc * 32);
                S0 = __builtin_amdgcn_wmma_f32_16x16x32_bf16(false, qf[dc], false, b0,
                                                             (short)0, S0, false, false);
                S1 = __builtin_amdgcn_wmma_f32_16x16x32_bf16(false, qf[dc], false, b1,
                                                             (short)0, S1, false, false);
            }
#pragma unroll
            for (int r = 0; r < 8; ++r) {
                const int qr_ = qrow0 + r + 8 * hi;
                float s0 = (kc0 + col      <= qr_) ? S0[r] * sm : -1e30f;
                float s1 = (kc0 + 16 + col <= qr_) ? S1[r] * sm : -1e30f;
                float mx = fmaxf(s0, s1);
                mx = fmaxf(mx, __shfl_xor(mx, 1));
                mx = fmaxf(mx, __shfl_xor(mx, 2));
                mx = fmaxf(mx, __shfl_xor(mx, 4));
                mx = fmaxf(mx, __shfl_xor(mx, 8));
                float mnew = fmaxf(mrow[r], mx);
                float corr = __expf(mrow[r] - mnew);
                float e0 = (s0 < -1e29f) ? 0.f : __expf(s0 - mnew);
                float e1 = (s1 < -1e29f) ? 0.f : __expf(s1 - mnew);
                float rs = e0 + e1;
                rs += __shfl_xor(rs, 1);
                rs += __shfl_xor(rs, 2);
                rs += __shfl_xor(rs, 4);
                rs += __shfl_xor(rs, 8);
                mrow[r] = mnew;
                lrow[r] = lrow[r] * corr + rs;
#pragma unroll
                for (int t = 0; t < 8; ++t) O[t][r] *= corr;
                const int m = r + 8 * hi;
                sP[wave][m * 32 + col]      = f2bf(e0);
                sP[wave][m * 32 + 16 + col] = f2bf(e1);
            }
            __syncthreads();
            v16bf pf = load_fragA(&sP[wave][0], 32, 0, 0);
            __syncthreads();
#pragma unroll
            for (int t = 0; t < 8; ++t) {
                v16bf vb = load_fragB(Vb, SEQ, t * 16, win * WINDOW + kc0);
                O[t] = __builtin_amdgcn_wmma_f32_16x16x32_bf16(false, pf, false, vb,
                                                               (short)0, O[t], false, false);
            }
        }
#pragma unroll
        for (int t = 0; t < 8; ++t) {
            int d = t * 16 + col;
#pragma unroll
            for (int r = 0; r < 8; ++r) {
                int s = win * WINDOW + qrow0 + r + 8 * hi;
                float inv = (lrow[r] > 0.f) ? 1.f / lrow[r] : 0.f;
                attn[(long)s * HIDDEN + head * HDIM + d] = f2bf(O[t][r] * inv);
            }
        }
    }
}

// Output projection GEMM -> f32.  grid = (SEQ/128, HIDDEN/128)  block = 256
__global__ __launch_bounds__(256) void out_gemm(
    const bf16* __restrict__ Ab, const bf16* __restrict__ Wob, float* __restrict__ out)
{
    __shared__ bf16 sA[2 * 128 * 32];
    __shared__ bf16 sB[2 * 128 * 32];
    const int m0 = blockIdx.x * 128;
    const int n0 = blockIdx.y * 128;
    const int tid = threadIdx.x, wave = tid >> 5, lane = tid & 31;

    v8f acc[8];
#pragma unroll
    for (int t = 0; t < 8; ++t)
#pragma unroll
        for (int e = 0; e < 8; ++e) acc[t][e] = 0.f;

    gemm128_async(Ab + (long)m0 * HIDDEN,
                  Wob + (long)n0 * HIDDEN,
                  sA, sB, acc);

    const int col = lane & 15, hi = lane >> 4;
#pragma unroll
    for (int t = 0; t < 8; ++t) {
        int n = n0 + t * 16 + col;
#pragma unroll
        for (int r = 0; r < 8; ++r) {
            int s = m0 + wave * 16 + hi * 8 + r;
            out[(long)s * HIDDEN + n] = acc[t][r];
        }
    }
}

extern "C" void kernel_launch(void* const* d_in, const int* in_sizes, int n_in,
                              void* d_out, int out_size, void* d_ws, size_t ws_size,
                              hipStream_t stream) {
    (void)in_sizes; (void)n_in; (void)out_size; (void)ws_size;
    const float* X   = (const float*)d_in[0];
    const int*   msk = (const int*)d_in[1];
    const int*   pos = (const int*)d_in[2];
    const float* Wq  = (const float*)d_in[3];
    const float* Wk  = (const float*)d_in[4];
    const float* Wv  = (const float*)d_in[5];
    const float* Wo  = (const float*)d_in[6];
    float* out = (float*)d_out;

    // Workspace layout (192 MiB total)
    char* ws = (char*)d_ws;
    bf16* Xb  = (bf16*)ws;  ws += (long)SEQ * HIDDEN * 2;        // 32 MiB
    bf16* Wqb = (bf16*)ws;  ws += (long)HIDDEN * HIDDEN * 2;     // 8 MiB  (Wq|Wk|Wv contiguous)
    bf16* Wkb = (bf16*)ws;  ws += (long)HIDDEN * HIDDEN * 2;
    bf16* Wvb = (bf16*)ws;  ws += (long)HIDDEN * HIDDEN * 2;
    bf16* Wob = (bf16*)ws;  ws += (long)HIDDEN * HIDDEN * 2;
    bf16* Qh  = (bf16*)ws;  ws += (long)NHEADS * SEQ * HDIM * 2; // 32 MiB
    bf16* Kh  = (bf16*)ws;  ws += (long)NHEADS * SEQ * HDIM * 2;
    bf16* Vt  = (bf16*)ws;  ws += (long)NHEADS * HDIM * SEQ * 2;
    bf16* At  = (bf16*)ws;  ws += (long)SEQ * HIDDEN * 2;

    f32_to_bf16_kernel<<<4096, 256, 0, stream>>>(X,  Xb,  (long)SEQ * HIDDEN / 4);
    f32_to_bf16_kernel<<<2048, 256, 0, stream>>>(Wq, Wqb, (long)HIDDEN * HIDDEN / 4);
    f32_to_bf16_kernel<<<2048, 256, 0, stream>>>(Wk, Wkb, (long)HIDDEN * HIDDEN / 4);
    f32_to_bf16_kernel<<<2048, 256, 0, stream>>>(Wv, Wvb, (long)HIDDEN * HIDDEN / 4);
    f32_to_bf16_kernel<<<2048, 256, 0, stream>>>(Wo, Wob, (long)HIDDEN * HIDDEN / 4);

    qkv_gemm<<<dim3(SEQ / 128, NHEADS, 3), 256, 0, stream>>>(Xb, Wqb, msk, pos, Qh, Kh, Vt);
    win_attn<<<dim3(NWIN, NHEADS), 256, 0, stream>>>(Qh, Kh, Vt, At);
    out_gemm<<<dim3(SEQ / 128, HIDDEN / 128), 256, 0, stream>>>(At, Wob, out);
}
